// ODconv2d_11991548691233
// MI455X (gfx1250) — compile-verified, matrix operand order checked by reading, not run on hardware
//
#include <hip/hip_runtime.h>

typedef __attribute__((ext_vector_type(16))) _Float16 v16h;
typedef __attribute__((ext_vector_type(8)))  _Float16 v8h;
typedef __attribute__((ext_vector_type(4)))  _Float16 v4h;
typedef __attribute__((ext_vector_type(8)))  float    v8f;

union AFrag { v16h v; v8h h8[2]; };

// ---------------- Kernel 1: global average pool  s[b,c] = mean(x[b,c,:,:]) ----
__global__ __launch_bounds__(256)
void k_gap(const float* __restrict__ x, float* __restrict__ s)
{
    const int bc = blockIdx.x;                 // b*C + c
    const float* xp = x + (size_t)bc * 3136;
    float acc = 0.f;
    for (int i = threadIdx.x; i < 3136; i += 256) acc += xp[i];
    __shared__ float red[256];
    red[threadIdx.x] = acc;
    __syncthreads();
    for (int off = 128; off > 0; off >>= 1) {
        if (threadIdx.x < off) red[threadIdx.x] += red[threadIdx.x + off];
        __syncthreads();
    }
    if (threadIdx.x == 0) s[bc] = red[0] * (1.f / 3136.f);
}

// ---------------- Kernel 2: attention trunk + 4 heads (one block per sample) --
__global__ __launch_bounds__(256)
void k_attn(const float* __restrict__ s,
            const float* __restrict__ fc_w, const float* __restrict__ bn_gamma,
            const float* __restrict__ bn_beta,
            const float* __restrict__ ch_w, const float* __restrict__ ch_b,
            const float* __restrict__ fl_w, const float* __restrict__ fl_b,
            const float* __restrict__ sp_w, const float* __restrict__ sp_b,
            const float* __restrict__ kn_w, const float* __restrict__ kn_b,
            float* __restrict__ ch, float* __restrict__ fl,
            float* __restrict__ sp, float* __restrict__ kn)
{
    constexpr int Cc = 256, Aa = 16;
    const int b = blockIdx.x;
    const int t = threadIdx.x;
    __shared__ float sS[Cc];
    __shared__ float sA[Aa];
    sS[t] = s[b * Cc + t];
    __syncthreads();
    if (t < Aa) {
        float acc = 0.f;
        for (int c = 0; c < Cc; ++c) acc += sS[c] * fc_w[t * Cc + c];
        acc = acc * (bn_gamma[t] * rsqrtf(1.f + 1e-5f)) + bn_beta[t];
        sA[t] = fmaxf(acc, 0.f);
    }
    __syncthreads();
    { // channel head (C == 256 == blockDim)
        float acc = ch_b[t];
        for (int j = 0; j < Aa; ++j) acc += sA[j] * ch_w[t * Aa + j];
        ch[b * Cc + t] = 1.f / (1.f + __expf(-acc));
    }
    { // filter head (O == 256 == blockDim)
        float acc = fl_b[t];
        for (int j = 0; j < Aa; ++j) acc += sA[j] * fl_w[t * Aa + j];
        fl[b * 256 + t] = 1.f / (1.f + __expf(-acc));
    }
    if (t < 9) { // spatial head
        float acc = sp_b[t];
        for (int j = 0; j < Aa; ++j) acc += sA[j] * sp_w[t * Aa + j];
        sp[b * 9 + t] = 1.f / (1.f + __expf(-acc));
    }
    if (t == 0) { // kernel head, softmax over K=4
        float l[4], m = -1e30f;
        for (int kq = 0; kq < 4; ++kq) {
            float acc = kn_b[kq];
            for (int j = 0; j < Aa; ++j) acc += sA[j] * kn_w[kq * Aa + j];
            l[kq] = acc; m = fmaxf(m, acc);
        }
        float sum = 0.f;
        for (int kq = 0; kq < 4; ++kq) { l[kq] = __expf(l[kq] - m); sum += l[kq]; }
        for (int kq = 0; kq < 4; ++kq) kn[b * 4 + kq] = l[kq] / sum;
    }
}

// ---------------- Kernel 3: xwH[b,c,hw] = f16( x * ch[b,c] ) ------------------
__global__ __launch_bounds__(256)
void k_xw(const float* __restrict__ x, const float* __restrict__ ch,
          _Float16* __restrict__ xwH)
{
    constexpr int HW = 3136, Cc = 256;
    const long long e4 = (long long)blockIdx.x * 256 + threadIdx.x;
    const long long e  = e4 * 4;
    if (e >= (long long)32 * Cc * HW) return;
    const int bc = (int)(e / HW);              // float4 never crosses a channel
    const float g = ch[bc];
    const float4 xv = *(const float4*)&x[e];
    v4h o;
    o[0] = (_Float16)(xv.x * g); o[1] = (_Float16)(xv.y * g);
    o[2] = (_Float16)(xv.z * g); o[3] = (_Float16)(xv.w * g);
    *(v4h*)&xwH[e] = o;
}

// ---------------- Kernel 4: aggH[b, o, c*9+p] = f16( sp * Σ_k kn * W ) --------
__global__ __launch_bounds__(256)
void k_agg(const float* __restrict__ weight, const float* __restrict__ sp,
           const float* __restrict__ kn, _Float16* __restrict__ aggH)
{
    constexpr long long PER = 256LL * 256 * 9;            // O*C*9
    const long long idx = (long long)blockIdx.x * 256 + threadIdx.x;
    if (idx >= 32 * PER) return;
    const int b = (int)(idx / PER);
    const long long r = idx - (long long)b * PER;         // (o*256+c)*9+p
    const int p = (int)(r % 9);
    float acc = 0.f;
#pragma unroll
    for (int kq = 0; kq < 4; ++kq)
        acc += kn[b * 4 + kq] * weight[(long long)kq * PER + r];
    aggH[idx] = (_Float16)(acc * sp[b * 9 + p]);
}

// ---------------- Kernel 5: implicit-GEMM conv with WMMA + async LDS copy -----
// Per sample b: D[O=256, HW=3136] = A[O, K=2304] x B[K, HW], fused *fl gate.
// 256 threads = 8 wave32, tile 128(M) x 64(N); each wave owns 32x32 = 2x2 WMMA
// tiles (4 v_wmma / K-step). A tile is staged with GLOBAL_LOAD_ASYNC_TO_LDS_B128
// into double-buffered LDS (ASYNCcnt-tracked, fully concurrent with WMMA);
// B is an im2col register gather stored to the other LDS buffer.
__global__ __launch_bounds__(256)
void k_conv_wmma(const _Float16* __restrict__ aggH,
                 const _Float16* __restrict__ xwH,
                 const float* __restrict__ fl,
                 float* __restrict__ out)
{
    constexpr int C = 256, HW = 3136, Wd = 56, KD = 2304, O = 256;
    const int b     = blockIdx.z;
    const int mBase = blockIdx.y * 128;
    const int nBase = blockIdx.x * 64;
    const int t     = threadIdx.x;
    const int lane  = t & 31;
    const int wid   = t >> 5;
    const int wm    = (wid & 3) * 32;          // wave M offset in tile
    const int wn    = (wid >> 2) * 32;         // wave N offset in tile

    __shared__ __align__(16) _Float16 As[2][128][40]; // double buffered, +8 pad
    __shared__ __align__(16) _Float16 Bs[2][64][40];  // transposed: [N][K]

    const _Float16* aggB = aggH + (size_t)b * (size_t)(O * KD);
    const _Float16* xwB  = xwH  + (size_t)b * (size_t)(C * HW);

    // A staging: 128 rows x 32 halfs = 512 16B-chunks; 2 async b128 per thread
    const int arow  = t >> 2;                  // 0..63  (and +64)
    const int akoff = (t & 3) * 8;             // 0,8,16,24
    // B im2col gather: 64 cols x 32 halfs = 512 4-half chunks; 2 per thread
    const int bn  = t >> 3;                    // 0..31  (and +32)
    const int bk  = (t & 7) * 4;               // 0..28
    const int nG0 = nBase + bn;
    const int nG1 = nG0 + 32;
    const int h0 = nG0 / Wd, w0 = nG0 % Wd;
    const int h1 = nG1 / Wd, w1 = nG1 % Wd;

    // ISA 16-bit fragment layout: lane<16 -> K[0..7]+K[16..23],
    // lane>=16 -> K[8..15]+K[24..31]; row/col = lane&15.
    const int half  = lane >> 4;
    const int row16 = lane & 15;
    const int kb    = half * 8;

    // per-lane global byte offsets of this thread's two A chunks (K-slice 0)
    const unsigned aOffG0 = (unsigned)(((mBase + arow)      * KD + akoff) * 2);
    const unsigned aOffG1 = (unsigned)(((mBase + arow + 64) * KD + akoff) * 2);

    // direct global->LDS copy of the two A chunks for K-slice kk into buf
    auto issueAsyncA = [&](int kk, int buf) {
        const unsigned lds0 = (unsigned)(uintptr_t)&As[buf][arow][akoff];
        const unsigned lds1 = (unsigned)(uintptr_t)&As[buf][arow + 64][akoff];
        const unsigned g0   = aOffG0 + (unsigned)(kk * 2);
        const unsigned g1   = aOffG1 + (unsigned)(kk * 2);
        asm volatile("global_load_async_to_lds_b128 %0, %1, %2\n\t"
                     "global_load_async_to_lds_b128 %3, %4, %2"
                     :: "v"(lds0), "v"(g0), "s"(aggB), "v"(lds1), "v"(g1)
                     : "memory");
    };

    auto gatherB = [&](int kk, int hh, int ww) -> v4h {
        v4h bv;
#pragma unroll
        for (int i = 0; i < 4; ++i) {
            const unsigned k = (unsigned)(kk + bk + i);
            const unsigned c = k / 9u;
            const unsigned p = k - c * 9u;
            const int ky = (int)(p / 3u) - 1;
            const int kx = (int)(p - (p / 3u) * 3u) - 1;
            const int h = hh + ky, w = ww + kx;
            _Float16 hv = (_Float16)0.f;
            if ((unsigned)h < 56u && (unsigned)w < 56u)
                hv = xwB[(size_t)c * HW + h * Wd + w];
            bv[i] = hv;
        }
        return bv;
    };

    v8f acc[2][2] = {};
    v4h rb0, rb1;

    // prologue: K-slice 0 into buffer 0
    issueAsyncA(0, 0);
    rb0 = gatherB(0, h0, w0);
    rb1 = gatherB(0, h1, w1);
    *(v4h*)&Bs[0][bn][bk]      = rb0;
    *(v4h*)&Bs[0][bn + 32][bk] = rb1;
    asm volatile("s_wait_asynccnt 0x0" ::: "memory");
    __syncthreads();

    for (int kk = 0; kk < KD; kk += 32) {
        const int  cur  = (kk >> 5) & 1;
        const int  nxt  = cur ^ 1;
        const bool more = (kk + 32) < KD;

        // launch next K-slice: async A copy + B gather (overlap with WMMA)
        if (more) {
            issueAsyncA(kk + 32, nxt);
            rb0 = gatherB(kk + 32, h0, w0);
            rb1 = gatherB(kk + 32, h1, w1);
        }

        AFrag fa[2], fb[2];
#pragma unroll
        for (int mt = 0; mt < 2; ++mt) {
            fa[mt].h8[0] = *(const v8h*)&As[cur][wm + mt * 16 + row16][kb];
            fa[mt].h8[1] = *(const v8h*)&As[cur][wm + mt * 16 + row16][kb + 16];
        }
#pragma unroll
        for (int nt = 0; nt < 2; ++nt) {
            fb[nt].h8[0] = *(const v8h*)&Bs[cur][wn + nt * 16 + row16][kb];
            fb[nt].h8[1] = *(const v8h*)&Bs[cur][wn + nt * 16 + row16][kb + 16];
        }
#pragma unroll
        for (int mt = 0; mt < 2; ++mt)
#pragma unroll
            for (int nt = 0; nt < 2; ++nt)
                acc[mt][nt] = __builtin_amdgcn_wmma_f32_16x16x32_f16(
                    false, fa[mt].v, false, fb[nt].v,
                    (short)0, acc[mt][nt], false, false);

        // retire next-slice staging, then make it visible to all waves
        if (more) {
            *(v4h*)&Bs[nxt][bn][bk]      = rb0;
            *(v4h*)&Bs[nxt][bn + 32][bk] = rb1;
            asm volatile("s_wait_asynccnt 0x0" ::: "memory");
        }
        __syncthreads();
    }

    // D layout: VGPR r -> M = r + 8*(lane>=16), N = lane&15. Fuse fl gate.
#pragma unroll
    for (int mt = 0; mt < 2; ++mt) {
#pragma unroll
        for (int nt = 0; nt < 2; ++nt) {
            const int nCol = nBase + wn + nt * 16 + row16;
#pragma unroll
            for (int r = 0; r < 8; ++r) {
                const int o = mBase + wm + mt * 16 + r + half * 8;
                out[((size_t)b * O + o) * HW + nCol] =
                    acc[mt][nt][r] * fl[b * O + o];
            }
        }
    }
}

// ---------------- host-side launch -------------------------------------------
extern "C" void kernel_launch(void* const* d_in, const int* in_sizes, int n_in,
                              void* d_out, int out_size, void* d_ws, size_t ws_size,
                              hipStream_t stream)
{
    const float* x      = (const float*)d_in[0];
    const float* weight = (const float*)d_in[1];
    const float* fc_w   = (const float*)d_in[2];
    const float* bn_g   = (const float*)d_in[3];
    const float* bn_b   = (const float*)d_in[4];
    const float* ch_w   = (const float*)d_in[5];
    const float* ch_b   = (const float*)d_in[6];
    const float* fl_w   = (const float*)d_in[7];
    const float* fl_b   = (const float*)d_in[8];
    const float* sp_w   = (const float*)d_in[9];
    const float* sp_b   = (const float*)d_in[10];
    const float* kn_w   = (const float*)d_in[11];
    const float* kn_b   = (const float*)d_in[12];
    float* out = (float*)d_out;

    // workspace layout (bytes), all 16B aligned
    char* ws = (char*)d_ws;
    float*    s_   = (float*)(ws + 0);                       // 32 KB
    float*    ch_  = (float*)(ws + 32768);                   // 32 KB
    float*    fl_  = (float*)(ws + 65536);                   // 32 KB
    float*    sp_  = (float*)(ws + 98304);                   // 1152 B
    float*    kn_  = (float*)(ws + 99456);                   // 512 B
    _Float16* xwH  = (_Float16*)(ws + 99968);                // 51,380,224 B
    _Float16* aggH = (_Float16*)(ws + 99968 + 51380224);     // 37,748,736 B

    k_gap <<<8192, 256, 0, stream>>>(x, s_);
    k_attn<<<32,   256, 0, stream>>>(s_, fc_w, bn_g, bn_b, ch_w, ch_b,
                                     fl_w, fl_b, sp_w, sp_b, kn_w, kn_b,
                                     ch_, fl_, sp_, kn_);
    k_xw  <<<25088, 256, 0, stream>>>(x, ch_, xwH);
    k_agg <<<73728, 256, 0, stream>>>(weight, sp_, kn_, aggH);
    dim3 g(49, 2, 32);  // N tiles (3136/64), M tiles (256/128), batch
    k_conv_wmma<<<g, 256, 0, stream>>>(aggH, xwH, fl_, out);
}
